// HandmadeLSTM_47837345743378
// MI455X (gfx1250) — matrix operand order repaired and checked by reading, not compiled, not run on hardware
//
#include <hip/hip_runtime.h>

// LSTM: T=1024, B=128, IN=512, H=512, 4H=2048
// d_out = [output (T*B*H) | h_final (B*H) | c_final (B*H)]  fp32

typedef __attribute__((ext_vector_type(16))) _Float16 v16h;
typedef __attribute__((ext_vector_type(8)))  _Float16 v8h;
typedef __attribute__((ext_vector_type(4)))  _Float16 v4h;
typedef __attribute__((ext_vector_type(8)))  float    v8f;
typedef __attribute__((ext_vector_type(4)))  float    v4f;
typedef __attribute__((ext_vector_type(4)))  int      v4i;

#define LSTM_T  1024
#define LSTM_B  128
#define LSTM_IN 512
#define LSTM_H  512
#define LSTM_G  2048            // 4*H
#define LSTM_K  1024            // IN + H (fused A = [x_t | h])
#define A_PAD   8               // halves of row padding in LDS

#define GLOBAL_AS __attribute__((address_space(1)))
#define LDS_AS    __attribute__((address_space(3)))

#if __has_builtin(__builtin_amdgcn_global_load_async_to_lds_b128) && \
    __has_builtin(__builtin_amdgcn_s_wait_asynccnt)
#define USE_ASYNC_LDS 1
#else
#define USE_ASYNC_LDS 0
#endif

// ---------------------------------------------------------------------------
// prep: Wcat[n][k] f16 (k<512 -> W_ih[n][k], else W_hh[n][k-512]),
//       bias = b_ih + b_hh, zero h (f16) and c (f32) state.
// ---------------------------------------------------------------------------
__global__ void lstm_prep_kernel(const float* __restrict__ W_ih,
                                 const float* __restrict__ W_hh,
                                 const float* __restrict__ b_ih,
                                 const float* __restrict__ b_hh,
                                 _Float16* __restrict__ Wcat,
                                 float* __restrict__ bias,
                                 _Float16* __restrict__ hbuf,
                                 float* __restrict__ cbuf) {
    int gid = blockIdx.x * blockDim.x + threadIdx.x;
    if (gid < LSTM_G * LSTM_K) {
        int n = gid >> 10;
        int k = gid & (LSTM_K - 1);
        float v = (k < LSTM_IN) ? W_ih[(size_t)n * LSTM_IN + k]
                                : W_hh[(size_t)n * LSTM_H + (k - LSTM_IN)];
        Wcat[gid] = (_Float16)v;
    }
    if (gid < LSTM_G) bias[gid] = b_ih[gid] + b_hh[gid];
    if (gid < LSTM_B * LSTM_H) {
        hbuf[gid] = (_Float16)0.0f;
        cbuf[gid] = 0.0f;
    }
}

// ---------------------------------------------------------------------------
// One time step. Grid: 256 blocks = 8 m-tiles (batch/16) x 32 n-tiles (H/16).
// Block: 256 threads = 8 waves. Wave w: gate = w&3, K-half = w>>2.
// z[m, g*512+n] = sum_k A[m,k] * Wcat[g*512+n, k],  A = [x_t | h_prev] (f16)
// Split-K partials reduced in LDS; fused gate math updates c, h.
// ---------------------------------------------------------------------------
__global__ __launch_bounds__(256)
void lstm_step_kernel(const float* __restrict__ x_t,      // [B, IN] fp32
                      const _Float16* __restrict__ Wcat,  // [2048, 1024] f16
                      const float* __restrict__ bias,     // [2048] fp32
                      _Float16* __restrict__ hbuf,        // [B, H] f16 (in/out)
                      float* __restrict__ cbuf,           // [B, H] fp32 (in/out)
                      float* __restrict__ out_t) {        // [B, H] fp32
    __shared__ __align__(16) _Float16 As[16][LSTM_K + A_PAD];
    __shared__ float Zs[8][16][16];

    const int tid  = threadIdx.x;
    const int mt   = blockIdx.x >> 5;   // batch tile   0..7
    const int nt   = blockIdx.x & 31;   // column tile  0..31

#if USE_ASYNC_LDS
    // h half of A: pure f16 global->LDS copy, use gfx1250 async-to-LDS DMA
    // (ASYNCcnt-tracked, no VGPR staging); overlaps with x conversion below.
    for (int idx = tid; idx < 16 * (LSTM_H / 8); idx += 256) {
        int m = idx >> 6, k8 = idx & 63;
        __builtin_amdgcn_global_load_async_to_lds_b128(
            (GLOBAL_AS v4i*)(hbuf + (size_t)(mt * 16 + m) * LSTM_H + k8 * 8),
            (LDS_AS v4i*)&As[m][LSTM_IN + k8 * 8], 0, 0);
    }
#endif

    // x half of A: vectorized f32 load -> f16 convert -> LDS (b64 stores)
    for (int idx = tid; idx < 16 * (LSTM_IN / 4); idx += 256) {
        int m = idx >> 7, k4 = idx & 127;
        v4f xv = *(const v4f*)(x_t + (size_t)(mt * 16 + m) * LSTM_IN + k4 * 4);
        v4h hv;
        hv.x = (_Float16)xv.x; hv.y = (_Float16)xv.y;
        hv.z = (_Float16)xv.z; hv.w = (_Float16)xv.w;
        *(v4h*)&As[m][k4 * 4] = hv;
    }

#if USE_ASYNC_LDS
    __builtin_amdgcn_s_wait_asynccnt(0);
#else
    for (int idx = tid; idx < 16 * (LSTM_H / 8); idx += 256) {
        int m = idx >> 6, k8 = idx & 63;
        *(v8h*)&As[m][LSTM_IN + k8 * 8] =
            *(const v8h*)(hbuf + (size_t)(mt * 16 + m) * LSTM_H + k8 * 8);
    }
#endif
    __syncthreads();

    const int wv   = tid >> 5;          // wave 0..7
    const int gate = wv & 3;            // 0:i 1:f 2:o 3:g
    const int kh   = wv >> 2;           // K half: 0 -> x part, 1 -> h part
    const int lane = tid & 31;
    const int ln   = lane & 15;         // row (A) / column (B) index
    const int hiw  = lane >> 4;         // half-wave select
    const int n0   = gate * LSTM_H + nt * 16;                  // z column base
    const _Float16* wrow = Wcat + (size_t)(n0 + ln) * LSTM_K + kh * 512;
    const _Float16* arow = &As[ln][kh * 512];

    v8f acc = {};
    #pragma unroll 4
    for (int k0 = 0; k0 < 512; k0 += 32) {
        // A frag (16-bit 16x32 layout): lanes0-15 K{0..7,16..23}, lanes16-31 +8
        v8h alo = *(const v8h*)(arow + k0 + hiw * 8);
        v8h ahi = *(const v8h*)(arow + k0 + 16 + hiw * 8);
        v16h a  = __builtin_shufflevector(alo, ahi,
                     0, 1, 2, 3, 4, 5, 6, 7, 8, 9, 10, 11, 12, 13, 14, 15);
        // B frag (32x16): same K order as A: lanes0-15 K{0..7,16..23}, lanes16-31 +8
        v8h blo = *(const v8h*)(wrow + k0 + hiw * 8);
        v8h bhi = *(const v8h*)(wrow + k0 + 16 + hiw * 8);
        v16h b  = __builtin_shufflevector(blo, bhi,
                     0, 1, 2, 3, 4, 5, 6, 7, 8, 9, 10, 11, 12, 13, 14, 15);
        acc = __builtin_amdgcn_wmma_f32_16x16x32_f16(
                  false, a, false, b, (short)0, acc, false, false);
    }

    // Spill C tile (VGPR r, lane<16 -> M=r, lane>=16 -> M=r+8; N=lane&15)
    #pragma unroll
    for (int r = 0; r < 8; ++r)
        Zs[wv][r + hiw * 8][ln] = acc[r];
    __syncthreads();

    // Fused split-K reduce + gate pointwise, 1 element/thread
    {
        int m = tid >> 4, n = tid & 15;
        int col = nt * 16 + n;
        float zi = Zs[0][m][n] + Zs[4][m][n] + bias[0 * LSTM_H + col];
        float zf = Zs[1][m][n] + Zs[5][m][n] + bias[1 * LSTM_H + col];
        float zo = Zs[2][m][n] + Zs[6][m][n] + bias[2 * LSTM_H + col];
        float zg = Zs[3][m][n] + Zs[7][m][n] + bias[3 * LSTM_H + col];
        float ig = 1.0f / (1.0f + expf(-zi));
        float fg = 1.0f / (1.0f + expf(-zf));
        float og = 1.0f / (1.0f + expf(-zo));
        float gg = tanhf(zg);
        size_t idx = (size_t)(mt * 16 + m) * LSTM_H + col;
        float c = fg * cbuf[idx] + ig * gg;
        cbuf[idx] = c;
        float h = og * tanhf(c);
        out_t[idx] = h;
        hbuf[idx] = (_Float16)h;
    }
}

// ---------------------------------------------------------------------------
// finalize: copy h_final (= output[T-1]) and c_final to the output tail.
// ---------------------------------------------------------------------------
__global__ void lstm_final_kernel(const float* __restrict__ last_h,
                                  const float* __restrict__ cbuf,
                                  float* __restrict__ out_tail) {
    int gid = blockIdx.x * blockDim.x + threadIdx.x;
    if (gid < LSTM_B * LSTM_H) {
        out_tail[gid]                   = last_h[gid];
        out_tail[LSTM_B * LSTM_H + gid] = cbuf[gid];
    }
}

// ---------------------------------------------------------------------------
extern "C" void kernel_launch(void* const* d_in, const int* in_sizes, int n_in,
                              void* d_out, int out_size, void* d_ws, size_t ws_size,
                              hipStream_t stream) {
    (void)in_sizes; (void)n_in; (void)out_size; (void)ws_size;
    const float* x    = (const float*)d_in[0];  // [T,B,IN]
    const float* W_ih = (const float*)d_in[1];  // [4H,IN]
    const float* b_ih = (const float*)d_in[2];  // [4H]
    const float* W_hh = (const float*)d_in[3];  // [4H,H]
    const float* b_hh = (const float*)d_in[4];  // [4H]
    float* out = (float*)d_out;

    // Workspace layout (~4.4 MiB total)
    char* ws = (char*)d_ws;
    _Float16* Wcat = (_Float16*)ws;                               // 4 MiB
    float*    bias = (float*)(ws + (size_t)4 * 1024 * 1024);      // 8 KiB
    _Float16* hbuf = (_Float16*)(ws + (size_t)4 * 1024 * 1024
                                    + 16 * 1024);                 // 128 KiB
    float*    cbuf = (float*)(ws + (size_t)4 * 1024 * 1024
                                 + 16 * 1024 + 128 * 1024);       // 256 KiB

    lstm_prep_kernel<<<(LSTM_G * LSTM_K + 255) / 256, 256, 0, stream>>>(
        W_ih, W_hh, b_ih, b_hh, Wcat, bias, hbuf, cbuf);

    for (int t = 0; t < LSTM_T; ++t) {
        lstm_step_kernel<<<256, 256, 0, stream>>>(
            x + (size_t)t * LSTM_B * LSTM_IN, Wcat, bias, hbuf, cbuf,
            out + (size_t)t * LSTM_B * LSTM_H);
    }

    lstm_final_kernel<<<(LSTM_B * LSTM_H + 255) / 256, 256, 0, stream>>>(
        out + (size_t)(LSTM_T - 1) * LSTM_B * LSTM_H, cbuf,
        out + (size_t)LSTM_T * LSTM_B * LSTM_H);
}
